// BDH_69638599737422
// MI455X (gfx1250) — compile-verified, matrix operand order checked by reading, not run on hardware
//
#include <hip/hip_runtime.h>
#include <hip/hip_bf16.h>
#include <math.h>

// ---------------------------------------------------------------------------
// BDH forward for MI455X (gfx1250, wave32, WMMA).
// Heavy GEMMs: v_wmma_f32_16x16x32_bf16, f32 accumulate, LDS double-buffered
// with CDNA5 async global->LDS copies (ASYNCcnt) when the builtin exists.
// ---------------------------------------------------------------------------

typedef __bf16  bf16_t;
typedef __bf16  v16bf __attribute__((ext_vector_type(16)));
typedef float   v8f   __attribute__((ext_vector_type(8)));

#define TWO_PI_F 6.283185307179586f

// ---------------- generic tiled transpose: f32 [R,C] -> bf16 [C,R] ----------
__global__ __launch_bounds__(256)
void transpose_bf16_kernel(const float* __restrict__ in, bf16_t* __restrict__ out,
                           int R, int C) {
  __shared__ float tile[32][33];
  const size_t bo = (size_t)blockIdx.z * (size_t)R * (size_t)C;
  in  += bo;
  out += bo;
  const int c0 = blockIdx.x * 32;
  const int r0 = blockIdx.y * 32;
  const int tx = threadIdx.x;  // 0..31
  const int ty = threadIdx.y;  // 0..7
#pragma unroll
  for (int i = 0; i < 32; i += 8)
    tile[ty + i][tx] = in[(size_t)(r0 + ty + i) * C + (c0 + tx)];
  __syncthreads();
#pragma unroll
  for (int i = 0; i < 32; i += 8)
    out[(size_t)(c0 + ty + i) * R + (r0 + tx)] = (bf16_t)tile[tx][ty + i];
}

// ---------------- block-wide LayerNorm helper (blockDim.x == 256) -----------
__device__ __forceinline__ float block_ln256(float v, float* red) {
  const int d = threadIdx.x;
  red[d] = v;
  __syncthreads();
#pragma unroll
  for (int s = 128; s > 0; s >>= 1) {
    if (d < s) red[d] += red[d + s];
    __syncthreads();
  }
  const float mean = red[0] * (1.0f / 256.0f);
  __syncthreads();
  const float c = v - mean;
  red[d] = c * c;
  __syncthreads();
#pragma unroll
  for (int s = 128; s > 0; s >>= 1) {
    if (d < s) red[d] += red[d + s];
    __syncthreads();
  }
  const float rstd = rsqrtf(red[0] * (1.0f / 256.0f) + 1e-5f);
  __syncthreads();
  return c * rstd;
}

__global__ __launch_bounds__(256)
void embed_ln_kernel(const int* __restrict__ idx, const float* __restrict__ embed,
                     float* __restrict__ x, bf16_t* __restrict__ xbf) {
  __shared__ float red[256];
  const int t = blockIdx.x;
  const int d = threadIdx.x;
  const float v = embed[(size_t)idx[t] * 256 + d];
  const float o = block_ln256(v, red);
  x[(size_t)t * 256 + d]   = o;
  xbf[(size_t)t * 256 + d] = (bf16_t)o;
}

__global__ __launch_bounds__(256)
void ln_rows_kernel(const float* __restrict__ in, bf16_t* __restrict__ out) {
  __shared__ float red[256];
  const size_t r = blockIdx.x;
  const int d = threadIdx.x;
  const float o = block_ln256(in[r * 256 + d], red);
  out[r * 256 + d] = (bf16_t)o;
}

__global__ __launch_bounds__(256)
void ln_add_ln_kernel(float* __restrict__ x, bf16_t* __restrict__ xbf,
                      const float* __restrict__ yraw) {
  __shared__ float red[256];
  const size_t t = blockIdx.x;
  const int d = threadIdx.x;
  const float yl = block_ln256(yraw[t * 256 + d], red);
  const float o  = block_ln256(x[t * 256 + d] + yl, red);
  x[t * 256 + d]   = o;
  xbf[t * 256 + d] = (bf16_t)o;
}

// RoPE: theta = 2^16 -> freq = 2^(-q/512)/2pi, computed on the fly.
__global__ __launch_bounds__(256)
void rope_kernel(const bf16_t* __restrict__ xs, bf16_t* __restrict__ qr) {
  const size_t p = (size_t)blockIdx.x * blockDim.x + threadIdx.x; // pair index
  const size_t per_t  = 4096;          // N/2
  const size_t per_ht = 512 * per_t;   // T * N/2
  const int h = (int)(p / per_ht);
  const size_t rem = p % per_ht;
  const int t = (int)(rem / per_t);
  const int i = (int)(rem % per_t);
  const int n0 = 2 * i;
  const float freq = exp2f(-(float)n0 * (1.0f / 512.0f)) * (1.0f / TWO_PI_F);
  float a = (float)t * freq;
  a -= floorf(a);
  a *= TWO_PI_F;
  float s, c;
  sincosf(a, &s, &c);
  const size_t base = (((size_t)h * 512 + t) * 8192) + n0;
  const float v0 = (float)xs[base];
  const float v1 = (float)xs[base + 1];
  qr[base]     = (bf16_t)(v0 * c - v1 * s);
  qr[base + 1] = (bf16_t)(v1 * c + v0 * s);
}

// ---------------------------------------------------------------------------
// WMMA GEMM:  C[M,N] = A[M,K] (bf16 row-major) * Bt[N,K]^T (bf16 row-major)
// 256 threads = 8 waves; block tile 128x256; K staged 32-wide through LDS,
// double-buffered via async global->LDS copies. Waves 2(M)x4(N), 4x4 frags.
// ---------------------------------------------------------------------------
#define BLK_M 128
#define BLK_N 256
#define KT    32
#define LDSK  40   // padded K stride (bf16 elems) to dodge bank conflicts

enum { EPI_F32 = 0, EPI_RELU_BF16 = 1, EPI_MASK_BF16 = 2, EPI_MUL_BF16 = 3 };

#if __has_builtin(__builtin_amdgcn_global_load_async_to_lds_b128)
#define BDH_ASYNC 1
#endif

// pointer types matching the builtin signature:
// arg0: int4 __device__* (addrspace 1), arg1: int4 __shared__* (addrspace 3)
typedef int v4i_t __attribute__((vector_size(16)));

// copy 16B global -> LDS (async on CDNA5, sync ds_store fallback)
__device__ __forceinline__ void cp_b128(const bf16_t* g, bf16_t* l) {
#ifdef BDH_ASYNC
  __builtin_amdgcn_global_load_async_to_lds_b128(
      (__attribute__((address_space(1))) v4i_t*)g,
      (__attribute__((address_space(3))) v4i_t*)l, 0, 0);
#else
  *(uint4*)l = *(const uint4*)g;
#endif
}

template <int Nw>
__device__ __forceinline__ void wait_async() {
#ifdef BDH_ASYNC
#if __has_builtin(__builtin_amdgcn_s_wait_asynccnt)
  __builtin_amdgcn_s_wait_asynccnt(Nw);
#else
  asm volatile("s_wait_asynccnt %0" :: "n"(Nw) : "memory");
#endif
#endif
}

// fragment load per ISA 7.12.2 (16-bit A/B): lane half l>>4 selects
// K blocks {kb..kb+7, kb+16..kb+23} with kb = 8*(l>>4).
__device__ __forceinline__ v16bf frag_ld(const bf16_t* base, int row0, int lane) {
  const bf16_t* p = base + (size_t)(row0 + (lane & 15)) * LDSK + ((lane >> 4) << 3);
  v16bf f;
#pragma unroll
  for (int i = 0; i < 8; ++i) {
    f[i]     = p[i];
    f[i + 8] = p[i + 16];
  }
  return f;
}

template <int EPI>
__global__ __launch_bounds__(256)
void wmma_gemm_bt(const bf16_t* __restrict__ A,  long long strideA,
                  const bf16_t* __restrict__ Bt, long long strideB,
                  void* __restrict__ Cv,         long long strideC, int ldc,
                  const bf16_t* __restrict__ X,  long long strideX, int ldx,
                  int M, int N, int K) {
  __shared__ bf16_t shA[2][BLK_M * LDSK];
  __shared__ bf16_t shB[2][BLK_N * LDSK];

  const int bz = blockIdx.z;
  A  += (size_t)bz * (size_t)strideA;
  Bt += (size_t)bz * (size_t)strideB;

  const int tid  = threadIdx.x;
  const int lane = tid & 31;
  const int w    = tid >> 5;     // wave 0..7
  const int wm   = w & 1;        // 2 waves along M (64 rows each)
  const int wn   = w >> 1;       // 4 waves along N (64 cols each)
  const int m0   = blockIdx.y * BLK_M;
  const int n0   = blockIdx.x * BLK_N;

  // D-fragment element coords: VGPR e -> M = e + 8*(lane>>4), N = lane&15
  const int colb = n0 + wn * 64 + (lane & 15);
  const int rowb = m0 + wm * 64 + ((lane >> 4) << 3);

  // Strictly-upper tiles of the causal score matrix are all-zero: skip K-loop.
  if (EPI == EPI_MASK_BF16 && n0 >= m0 + BLK_M) {
#pragma unroll
    for (int i = 0; i < 4; ++i)
#pragma unroll
      for (int j = 0; j < 4; ++j)
#pragma unroll
        for (int e = 0; e < 8; ++e) {
          const size_t offc = (size_t)bz * (size_t)strideC +
                              (size_t)(rowb + i * 16 + e) * ldc + (colb + j * 16);
          ((bf16_t*)Cv)[offc] = (bf16_t)0.0f;
        }
    return;
  }

  v8f acc[4][4];
#pragma unroll
  for (int i = 0; i < 4; ++i)
#pragma unroll
    for (int j = 0; j < 4; ++j)
#pragma unroll
      for (int e = 0; e < 8; ++e) acc[i][j][e] = 0.0f;

  const int ra = tid >> 1;   // A tile row handled by this thread (0..127)
  const int ha = tid & 1;    // which 32B half of the 64B A row slab
  const int nk = K / KT;

  auto issue = [&](int kt, int buf) {
    const int k0 = kt * KT;
    // A tile: 128 rows x 64B, 2 x b128 per thread
    const bf16_t* ga = A + (size_t)(m0 + ra) * K + k0 + ha * 16;
    bf16_t* la = &shA[buf][ra * LDSK + ha * 16];
    cp_b128(ga, la);
    cp_b128(ga + 8, la + 8);
    // B tile: 256 rows x 64B, 4 x b128 per thread (one full row)
    const bf16_t* gb = Bt + (size_t)(n0 + tid) * K + k0;
    bf16_t* lb = &shB[buf][tid * LDSK];
#pragma unroll
    for (int q = 0; q < 4; ++q) cp_b128(gb + q * 8, lb + q * 8);
  };

  issue(0, 0);
  for (int kt = 0; kt < nk; ++kt) {
    const int cur = kt & 1;
    if (kt + 1 < nk) {
      issue(kt + 1, cur ^ 1);   // stream next slab while computing current
      wait_async<6>();          // this stage's 6 copies landed (in-order)
    } else {
      wait_async<0>();
    }
    __syncthreads();

    v16bf af[4], bfg[4];
#pragma unroll
    for (int i = 0; i < 4; ++i) af[i]  = frag_ld(&shA[cur][0], wm * 64 + i * 16, lane);
#pragma unroll
    for (int j = 0; j < 4; ++j) bfg[j] = frag_ld(&shB[cur][0], wn * 64 + j * 16, lane);

#pragma unroll
    for (int i = 0; i < 4; ++i)
#pragma unroll
      for (int j = 0; j < 4; ++j)
        acc[i][j] = __builtin_amdgcn_wmma_f32_16x16x32_bf16(
            false, af[i], false, bfg[j], (short)0, acc[i][j], false, false);
    __syncthreads();
  }

  // epilogue
#pragma unroll
  for (int i = 0; i < 4; ++i) {
#pragma unroll
    for (int j = 0; j < 4; ++j) {
#pragma unroll
      for (int e = 0; e < 8; ++e) {
        const int row = rowb + i * 16 + e;
        const int col = colb + j * 16;
        float v = acc[i][j][e];
        const size_t off = (size_t)bz * (size_t)strideC + (size_t)row * ldc + col;
        if (EPI == EPI_F32) {
          ((float*)Cv)[off] = v;
        } else if (EPI == EPI_RELU_BF16) {
          ((bf16_t*)Cv)[off] = (bf16_t)fmaxf(v, 0.0f);
        } else if (EPI == EPI_MASK_BF16) {
          ((bf16_t*)Cv)[off] = (bf16_t)((col < row) ? v : 0.0f);  // strict lower
        } else { // EPI_MUL_BF16: relu(v) * x_sparse
          const float xv = (float)X[(size_t)bz * (size_t)strideX + (size_t)row * ldx + col];
          ((bf16_t*)Cv)[off] = (bf16_t)(fmaxf(v, 0.0f) * xv);
        }
      }
    }
  }
}

// ---------------------------------------------------------------------------
extern "C" void kernel_launch(void* const* d_in, const int* in_sizes, int n_in,
                              void* d_out, int out_size, void* d_ws, size_t ws_size,
                              hipStream_t stream) {
  (void)in_sizes; (void)n_in; (void)out_size; (void)ws_size;

  const int*   idx   = (const int*)d_in[0];
  const float* embed = (const float*)d_in[1];
  const float* enc   = (const float*)d_in[2];   // [4,256,8192]
  const float* encv  = (const float*)d_in[3];   // [4,256,8192]
  const float* dec   = (const float*)d_in[4];   // [32768,256]
  const float* lmh   = (const float*)d_in[5];   // [256,256]
  float* logits = (float*)d_out;                // [512,256]

  const int T = 512, D = 256, NH = 4, NN = 8192, V = 256;

  // --- workspace bump allocator (~122 MB total) ---
  char* base = (char*)d_ws;
  size_t off = 0;
  auto bump = [&](size_t bytes) -> void* {
    void* p = base + off;
    off += (bytes + 255) & ~(size_t)255;
    return p;
  };
  bf16_t* encT   = (bf16_t*)bump((size_t)NH * NN * D * 2);  // [h][N][D]
  bf16_t* encvT  = (bf16_t*)bump((size_t)NH * NN * D * 2);  // [h][N][D]
  bf16_t* decT   = (bf16_t*)bump((size_t)D * NH * NN * 2);  // [D][nh*N]
  bf16_t* lmT    = (bf16_t*)bump((size_t)V * D * 2);        // [V][D]
  float*  x      = (float*) bump((size_t)T * D * 4);
  bf16_t* xbf    = (bf16_t*)bump((size_t)T * D * 2);
  bf16_t* xbfT   = (bf16_t*)bump((size_t)D * T * 2);        // [D][T]
  bf16_t* xs     = (bf16_t*)bump((size_t)NH * T * NN * 2);  // x_sparse
  bf16_t* qr     = (bf16_t*)bump((size_t)NH * T * NN * 2);  // rotated; reused as xy
  bf16_t* scores = (bf16_t*)bump((size_t)NH * T * T * 2);
  float*  yKV    = (float*) bump((size_t)NH * T * D * 4);
  bf16_t* yKVbf  = (bf16_t*)bump((size_t)NH * T * D * 2);
  float*  yraw   = (float*) bump((size_t)T * D * 4);
  bf16_t* xy     = qr;  // qr dead after scores GEMM; reuse 32 MB

  const dim3 tb(32, 8);

  // one-time weight conversion/transposition to bf16, K-contiguous layouts
  transpose_bf16_kernel<<<dim3(NN / 32, D / 32, NH), tb, 0, stream>>>(enc,  encT,  D, NN);
  transpose_bf16_kernel<<<dim3(NN / 32, D / 32, NH), tb, 0, stream>>>(encv, encvT, D, NN);
  transpose_bf16_kernel<<<dim3(D / 32, (NH * NN) / 32, 1), tb, 0, stream>>>(dec, decT, NH * NN, D);
  transpose_bf16_kernel<<<dim3(V / 32, D / 32, 1), tb, 0, stream>>>(lmh, lmT, D, V);

  embed_ln_kernel<<<T, 256, 0, stream>>>(idx, embed, x, xbf);
  transpose_bf16_kernel<<<dim3(D / 32, T / 32, 1), tb, 0, stream>>>(x, xbfT, T, D);

  for (int layer = 0; layer < 6; ++layer) {
    // x_sparse = relu(X @ enc[h])            [h][T][N] bf16
    wmma_gemm_bt<EPI_RELU_BF16><<<dim3(NN / BLK_N, T / BLK_M, NH), 256, 0, stream>>>(
        xbf, 0, encT, (long long)NN * D, xs, (long long)T * NN, NN,
        nullptr, 0, 0, T, NN, D);
    // qr = RoPE(x_sparse)
    rope_kernel<<<(NH * T * (NN / 2)) / 256, 256, 0, stream>>>(xs, qr);
    // scores = tril(qr @ qr^T, -1)           [h][T][T] bf16
    wmma_gemm_bt<EPI_MASK_BF16><<<dim3(T / BLK_N, T / BLK_M, NH), 256, 0, stream>>>(
        qr, (long long)T * NN, qr, (long long)T * NN, scores, (long long)T * T, T,
        nullptr, 0, 0, T, T, NN);
    // yKV_raw = scores @ X                   [h][T][D] f32   (B^T = x^T = xbfT)
    wmma_gemm_bt<EPI_F32><<<dim3(D / BLK_N, T / BLK_M, NH), 256, 0, stream>>>(
        scores, (long long)T * T, xbfT, 0, yKV, (long long)T * D, D,
        nullptr, 0, 0, T, D, T);
    // yKV = ln(yKV_raw) -> bf16
    ln_rows_kernel<<<NH * T, 256, 0, stream>>>(yKV, yKVbf);
    // xy[t][h*N+n] = relu(yKV @ encv[h]) * x_sparse
    wmma_gemm_bt<EPI_MUL_BF16><<<dim3(NN / BLK_N, T / BLK_M, NH), 256, 0, stream>>>(
        yKVbf, (long long)T * D, encvT, (long long)NN * D, xy, (long long)NN, NH * NN,
        xs, (long long)T * NN, NN, T, NN, D);
    // y_raw = xy @ decoder                   [T][D] f32
    wmma_gemm_bt<EPI_F32><<<dim3(D / BLK_N, T / BLK_M, 1), 256, 0, stream>>>(
        xy, 0, decT, 0, yraw, 0, D,
        nullptr, 0, 0, T, D, NH * NN);
    // x = ln(x + ln(y_raw))
    ln_add_ln_kernel<<<T, 256, 0, stream>>>(x, xbf, yraw);
    transpose_bf16_kernel<<<dim3(D / 32, T / 32, 1), tb, 0, stream>>>(x, xbfT, T, D);
  }

  // logits = x @ lm_head                     [T][V] f32 -> d_out
  wmma_gemm_bt<EPI_F32><<<dim3(V / BLK_N, T / BLK_M, 1), 256, 0, stream>>>(
      xbf, 0, lmT, 0, logits, 0, V,
      nullptr, 0, 0, T, V, D);
}
